// RNNModel_35940286333547
// MI455X (gfx1250) — matrix operand order repaired and checked by reading, not compiled, not run on hardware
//
#include <hip/hip_runtime.h>

typedef __bf16 bf16;
typedef __attribute__((ext_vector_type(16))) __bf16 v16bf;
typedef __attribute__((ext_vector_type(8)))  __bf16 v8bf;
typedef __attribute__((ext_vector_type(8)))  float  v8f;

#define B_    64
#define T_    2048
#define IN_   128
#define H_    256
#define BLOCK 512                 // 16 waves of 32
#define RING  64                  // ring depth (steps) per batch tile
#define TILE_ELEMS (16*H_)        // 4096 bf16 per h1 step-tile (8 KB)

// ---- workspace layout (bytes) ----
#define RING_BYTES ((size_t)4 * RING * TILE_ELEMS * 2)   // 2 MiB
#define FLAGS_OFF  RING_BYTES                            // int prod[4] @+0, int cons[4] @+64

// ---- LDS layouts (bytes) ----
// layer 0 block:
#define L0_WIH0 0
#define L0_WHH0 (L0_WIH0 + H_*IN_*2)       // 65536
#define L0_H1   (L0_WHH0 + H_*H_*2)        // 196608 : [2][4096] bf16 ping-pong
// layer 1 block:
#define L1_WIH1 0
#define L1_WHH1 (L1_WIH1 + H_*H_*2)        // 131072
#define L1_H2   (L1_WHH1 + H_*H_*2)        // 262144 : [2][4096] bf16 ping-pong
#define L1_H1L  (L1_H2 + 2*TILE_ELEMS*2)   // 278528 : [4096] bf16 staging
#define L1_LAST (L1_H1L + TILE_ELEMS*2)    // 286720 : [4096] f32 final hidden
#define LDS_BYTES (L1_LAST + TILE_ELEMS*4) // 303104 (296 KB < 320 KB/WGP)

__device__ __forceinline__ v16bf combine(v8bf lo, v8bf hi) {
    v16bf r;
#pragma unroll
    for (int i = 0; i < 8; ++i) { r[i] = lo[i]; r[8 + i] = hi[i]; }
    return r;
}
// B fragment: B[k][n] = W[n][k]; lane owns col n, K chunks (half*8, 16+half*8)
__device__ __forceinline__ v16bf fragB(const bf16* W, int ncols, int n, int ks, int half) {
    const bf16* pb = W + (size_t)n * ncols + ks * 32 + half * 8;
    return combine(*(const v8bf*)pb, *(const v8bf*)(pb + 16));
}
// A fragment from a row-major 16xH bf16 tile: lane = row, K chunks (half*8, 16+half*8)
__device__ __forceinline__ v16bf fragA(const bf16* hbuf, int l16, int ks, int half) {
    const bf16* pa = hbuf + l16 * H_ + ks * 32 + half * 8;
    return combine(*(const v8bf*)pa, *(const v8bf*)(pa + 16));
}

__global__ void init_flags(int* flags) {
    if (threadIdx.x < 32) flags[threadIdx.x] = 0;   // prod[4] + pad + cons[4] + pad
}

__global__ void __launch_bounds__(BLOCK, 1)
rnn2_pipeline(const float* __restrict__ x,
              const float* __restrict__ Wih0, const float* __restrict__ Whh0,
              const float* __restrict__ bih0, const float* __restrict__ bhh0,
              const float* __restrict__ Wih1, const float* __restrict__ Whh1,
              const float* __restrict__ bih1, const float* __restrict__ bhh1,
              const float* __restrict__ Wfc,  const float* __restrict__ bfc,
              float* __restrict__ out, char* __restrict__ ws)
{
    extern __shared__ char lds[];
    int* prod = (int*)(ws + FLAGS_OFF);        // producer progress per batch tile
    int* cons = (int*)(ws + FLAGS_OFF + 64);   // consumer progress per batch tile

    const int tid   = threadIdx.x;
    const int wave  = tid >> 5;
    const int lane  = tid & 31;
    const int half  = lane >> 4;
    const int l16   = lane & 15;
    const int n     = wave * 16 + l16;         // output column owned by this lane
    const int p     = blockIdx.x & 3;          // batch tile 0..3 (rows p*16..p*16+15)
    const int layer = blockIdx.x >> 2;         // 0: producer, 1: consumer
    const int brow0 = p * 16;
    bf16* ring = (bf16*)ws + (size_t)p * RING * TILE_ELEMS;

    if (layer == 0) {
        // =========================== layer-0 producer ===========================
        bf16* wih0 = (bf16*)(lds + L0_WIH0);
        bf16* whh0 = (bf16*)(lds + L0_WHH0);
        bf16* h1b  = (bf16*)(lds + L0_H1);
        for (int i = tid; i < H_ * IN_; i += BLOCK) wih0[i] = (bf16)Wih0[i];
        for (int i = tid; i < H_ * H_;  i += BLOCK) whh0[i] = (bf16)Whh0[i];
        for (int i = tid; i < TILE_ELEMS; i += BLOCK) h1b[i] = (bf16)0.0f;
        __syncthreads();

        const float bias0 = bih0[n] + bhh0[n];
        const float* xrow = x + (size_t)(brow0 + l16) * T_ * IN_;
        int cur = 0;
#pragma unroll 1
        for (int t = 0; t < T_; ++t) {
            const bf16* h1c = h1b + cur * TILE_ELEMS;
            bf16*       h1n = h1b + (cur ^ 1) * TILE_ELEMS;
            const float* xt = xrow + (size_t)t * IN_;
            if (t + 1 < T_) __builtin_prefetch(xt + IN_, 0, 0);

            v8f acc;
#pragma unroll
            for (int i = 0; i < 8; ++i) acc[i] = bias0;

            // x_t @ Wih0^T : K = 128 (A from global f32, converted)
#pragma unroll
            for (int ks = 0; ks < 4; ++ks) {
                const float* pa = xt + ks * 32 + half * 8;
                v16bf a;
#pragma unroll
                for (int i = 0; i < 8; ++i) { a[i] = (bf16)pa[i]; a[8 + i] = (bf16)pa[16 + i]; }
                acc = __builtin_amdgcn_wmma_f32_16x16x32_bf16(false, a, false,
                        fragB(wih0, IN_, n, ks, half), (short)0, acc, false, false);
            }
            // h1 @ Whh0^T : K = 256 (A from LDS ping buffer)
#pragma unroll
            for (int ks = 0; ks < 8; ++ks) {
                acc = __builtin_amdgcn_wmma_f32_16x16x32_bf16(false,
                        fragA(h1c, l16, ks, half), false,
                        fragB(whh0, H_, n, ks, half), (short)0, acc, false, false);
            }
            // tanh -> h1[next] in LDS (C layout: VGPR i -> row i + 8*half, col n)
#pragma unroll
            for (int i = 0; i < 8; ++i)
                h1n[(i + 8 * half) * H_ + n] = (bf16)tanhf(acc[i]);
            __syncthreads();

            // back-pressure: don't overwrite a slot the consumer hasn't read
            if (t >= RING) {
                while (__hip_atomic_load(&cons[p], __ATOMIC_ACQUIRE,
                                         __HIP_MEMORY_SCOPE_AGENT) < t - RING + 1)
                    __builtin_amdgcn_s_sleep(2);
            }
            // publish h1[t] tile: LDS -> ring (coalesced 16B per thread)
            {
                v8bf* dst = (v8bf*)(ring + (size_t)(t & (RING - 1)) * TILE_ELEMS);
                dst[tid] = ((const v8bf*)h1n)[tid];
            }
            __threadfence();          // make ring stores visible at device scope
            __syncthreads();          // all threads' fences done
            if (tid == 0)
                __hip_atomic_store(&prod[p], t + 1, __ATOMIC_RELEASE,
                                   __HIP_MEMORY_SCOPE_AGENT);
            cur ^= 1;
        }
    } else {
        // =========================== layer-1 consumer ===========================
        bf16*  wih1 = (bf16*)(lds + L1_WIH1);
        bf16*  whh1 = (bf16*)(lds + L1_WHH1);
        bf16*  h2b  = (bf16*)(lds + L1_H2);
        bf16*  h1l  = (bf16*)(lds + L1_H1L);
        float* last = (float*)(lds + L1_LAST);
        for (int i = tid; i < H_ * H_; i += BLOCK) { wih1[i] = (bf16)Wih1[i]; whh1[i] = (bf16)Whh1[i]; }
        for (int i = tid; i < TILE_ELEMS; i += BLOCK) h2b[i] = (bf16)0.0f;
        __syncthreads();

        const float bias1 = bih1[n] + bhh1[n];
        int cur = 0;
#pragma unroll 1
        for (int t = 0; t < T_; ++t) {
            const bf16* h2c = h2b + cur * TILE_ELEMS;
            bf16*       h2n = h2b + (cur ^ 1) * TILE_ELEMS;

            // wait for h1[t]; acquire orders/refreshes the subsequent ring loads
            while (__hip_atomic_load(&prod[p], __ATOMIC_ACQUIRE,
                                     __HIP_MEMORY_SCOPE_AGENT) < t + 1)
                __builtin_amdgcn_s_sleep(2);

            // stage ring slot into LDS once (coalesced 16B per thread)
            {
                const v8bf* src = (const v8bf*)(ring + (size_t)(t & (RING - 1)) * TILE_ELEMS);
                ((v8bf*)h1l)[tid] = src[tid];
            }
            __syncthreads();

            v8f acc;
#pragma unroll
            for (int i = 0; i < 8; ++i) acc[i] = bias1;
            // h1_t @ Wih1^T : K = 256
#pragma unroll
            for (int ks = 0; ks < 8; ++ks) {
                acc = __builtin_amdgcn_wmma_f32_16x16x32_bf16(false,
                        fragA(h1l, l16, ks, half), false,
                        fragB(wih1, H_, n, ks, half), (short)0, acc, false, false);
            }
            // h2 @ Whh1^T : K = 256
#pragma unroll
            for (int ks = 0; ks < 8; ++ks) {
                acc = __builtin_amdgcn_wmma_f32_16x16x32_bf16(false,
                        fragA(h2c, l16, ks, half), false,
                        fragB(whh1, H_, n, ks, half), (short)0, acc, false, false);
            }
#pragma unroll
            for (int i = 0; i < 8; ++i) {
                float v = tanhf(acc[i]);
                h2n[(i + 8 * half) * H_ + n] = (bf16)v;
                if (t == T_ - 1) last[(i + 8 * half) * H_ + n] = v;
            }
            __syncthreads();
            if (tid == 0)
                __hip_atomic_store(&cons[p], t + 1, __ATOMIC_RELEASE,
                                   __HIP_MEMORY_SCOPE_AGENT);
            cur ^= 1;
        }

        // fused FC epilogue: out[b] = dot(last[b,:], Wfc) + bfc
        if (tid < 16) {
            float s = bfc[0];
            const float* lr = last + tid * H_;
            for (int c = 0; c < H_; ++c) s += lr[c] * Wfc[c];
            out[brow0 + tid] = s;
        }
    }
}

extern "C" void kernel_launch(void* const* d_in, const int* in_sizes, int n_in,
                              void* d_out, int out_size, void* d_ws, size_t ws_size,
                              hipStream_t stream) {
    (void)in_sizes; (void)n_in; (void)ws_size; (void)out_size;
    const float* x    = (const float*)d_in[0];
    const float* Wih0 = (const float*)d_in[1];
    const float* Whh0 = (const float*)d_in[2];
    const float* bih0 = (const float*)d_in[3];
    const float* bhh0 = (const float*)d_in[4];
    const float* Wih1 = (const float*)d_in[5];
    const float* Whh1 = (const float*)d_in[6];
    const float* bih1 = (const float*)d_in[7];
    const float* bhh1 = (const float*)d_in[8];
    const float* Wfc  = (const float*)d_in[9];
    const float* bfc  = (const float*)d_in[10];

    // reset producer/consumer counters every launch (graph-replay deterministic)
    init_flags<<<1, 32, 0, stream>>>((int*)((char*)d_ws + FLAGS_OFF));

    // 8 co-resident workgroups: blocks 0-3 = layer 0, blocks 4-7 = layer 1
    rnn2_pipeline<<<dim3(8), dim3(BLOCK), LDS_BYTES, stream>>>(
        x, Wih0, Whh0, bih0, bhh0, Wih1, Whh1, bih1, bhh1, Wfc, bfc,
        (float*)d_out, (char*)d_ws);
}
